// GraphTransformer_16870631539208
// MI455X (gfx1250) — matrix operand order repaired and checked by reading, not compiled, not run on hardware
//
#include <hip/hip_runtime.h>
#include <math.h>

// ---------------------------------------------------------------------------
// Problem constants (match reference)
// ---------------------------------------------------------------------------
#define NN    32768
#define EE    327680
#define GG    64
#define IN_F  771
#define HH    256
#define EDIM  4
#define LL    4
#define NHH   8
#define DD    32

// ---------------------------------------------------------------------------
// WMMA types / helpers (gfx1250, wave32)
// ---------------------------------------------------------------------------
typedef __attribute__((ext_vector_type(16))) __bf16          v16bf;
typedef __attribute__((ext_vector_type(8)))  unsigned int    v8u;
typedef __attribute__((ext_vector_type(8)))  float           v8f;

union Frag { v8u u; v16bf b; };

__device__ __forceinline__ unsigned short f2bf(float f) {
  unsigned int u = __float_as_uint(f);
  u += 0x7FFFu + ((u >> 16) & 1u);   // round-to-nearest-even
  return (unsigned short)(u >> 16);
}
__device__ __forceinline__ unsigned int pack2bf(float a, float b) {
  return (unsigned int)f2bf(a) | ((unsigned int)f2bf(b) << 16);
}

// 4 floats with only 4-byte alignment guarantee (K=771 rows are not 16B-aligned)
struct __align__(4) f4 { float x, y, z, w; };

// ---------------------------------------------------------------------------
// WMMA GEMM: C[M,Nc] = (relu)(A[M,Kc] @ B[Kc,Nc] + bias)
// Block = 256 threads = 8 waves; tile 64x128, K stepped by 32.
// Wave (wr,wc): rows wr*16..+16, cols wc*64..+64 -> 1 A frag x 4 B frags
// -> 4 x v_wmma_f32_16x16x32_bf16 per K-step.
// LDS: A row-major [64][40] (padded), B TRANSPOSED [128][40] so each lane's
// fragment is two contiguous 16B chunks (ds_load_b128), bank-conflict-free.
// ---------------------------------------------------------------------------
#define TILE_M 64
#define TILE_N 128
#define TILE_K 32
#define LDSPAD 40   // ushorts per row: 80B, multiple of 16B, 20-bank row skew

__global__ __launch_bounds__(256)
void gemm_wmma_bf16(const float* __restrict__ A, const float* __restrict__ B,
                    const float* __restrict__ bias, float* __restrict__ C,
                    int M, int Nc, int Kc, int relu)
{
  __shared__ __align__(16) unsigned short As[TILE_M][LDSPAD];
  __shared__ __align__(16) unsigned short Bs[TILE_N][LDSPAD];  // [n][k] transposed

  const int tid  = threadIdx.x;
  const int wave = tid >> 5;
  const int lane = tid & 31;
  const int wr   = wave & 3;     // row subtile (16 rows)
  const int wc   = wave >> 2;    // col subtile (64 cols)
  const int m0   = blockIdx.y * TILE_M;
  const int n0   = blockIdx.x * TILE_N;

  const int l15  = lane & 15;
  const int hi16 = lane >> 4;    // 0 / 1

  v8f acc0 = {}, acc1 = {}, acc2 = {}, acc3 = {};

  for (int k0 = 0; k0 < Kc; k0 += TILE_K) {
    const bool fast = (m0 + TILE_M <= M) && (n0 + TILE_N <= Nc) && (k0 + TILE_K <= Kc);
    if (fast) {
      // ---- A tile: 64x32 floats, 2 x f4 per thread, coalesced ----
#pragma unroll
      for (int i = 0; i < 2; ++i) {
        int idx = tid + i * 256;          // 0..511
        int r = idx >> 3, c4 = idx & 7;
        f4 v = *(const f4*)&A[(size_t)(m0 + r) * Kc + k0 + c4 * 4];
        unsigned int p0 = pack2bf(v.x, v.y), p1 = pack2bf(v.z, v.w);
        *(uint2*)&As[r][c4 * 4] = make_uint2(p0, p1);
      }
      // ---- B tile: 32x128 floats, 4 x f4 per thread, store transposed ----
#pragma unroll
      for (int i = 0; i < 4; ++i) {
        int idx = tid + i * 256;          // 0..1023
        int k = idx >> 5, c4 = idx & 31;
        f4 v = *(const f4*)&B[(size_t)(k0 + k) * Nc + n0 + c4 * 4];
        Bs[c4 * 4 + 0][k] = f2bf(v.x);
        Bs[c4 * 4 + 1][k] = f2bf(v.y);
        Bs[c4 * 4 + 2][k] = f2bf(v.z);
        Bs[c4 * 4 + 3][k] = f2bf(v.w);
      }
      // prefetch next K tile
      if (k0 + TILE_K < Kc)
        __builtin_prefetch(&A[(size_t)(m0 + (tid >> 3)) * Kc + (k0 + TILE_K) + (tid & 7) * 4], 0, 1);
      if (k0 + 2 * TILE_K <= Kc)
        __builtin_prefetch(&B[(size_t)(k0 + TILE_K + (tid >> 5)) * Nc + n0 + (tid & 31) * 4], 0, 1);
    } else {
      // ---- bounds-checked path (K tail / tiny-M precompute) ----
      for (int idx = tid; idx < TILE_M * TILE_K; idx += 256) {
        int r = idx >> 5, c = idx & 31;
        int gm = m0 + r, gk = k0 + c;
        float v = (gm < M && gk < Kc) ? A[(size_t)gm * Kc + gk] : 0.f;
        As[r][c] = f2bf(v);
      }
      for (int idx = tid; idx < TILE_K * TILE_N; idx += 256) {
        int r = idx >> 7, c = idx & 127;
        int gk = k0 + r, gn = n0 + c;
        float v = (gk < Kc && gn < Nc) ? B[(size_t)gk * Nc + gn] : 0.f;
        Bs[c][r] = f2bf(v);
      }
    }
    __syncthreads();

    // A fragment: lane -> row wr*16 + (l&15); K = hi16*8 + {0..7, 16..23}
    const uint4* arp = (const uint4*)As[wr * 16 + l15];
    uint4 alo = arp[hi16];          // ushorts hi16*8 .. +7
    uint4 ahi = arp[hi16 + 2];      // ushorts hi16*8+16 .. +7
    Frag af;
    af.u[0] = alo.x; af.u[1] = alo.y; af.u[2] = alo.z; af.u[3] = alo.w;
    af.u[4] = ahi.x; af.u[5] = ahi.y; af.u[6] = ahi.z; af.u[7] = ahi.w;

    // B fragments: lane -> col wc*64 + t*16 + (l&15); K = hi16*8 + {0..7, 16..23}
#define LOAD_BFRAG(FR, T)                                                \
    {                                                                    \
      const uint4* brp = (const uint4*)Bs[wc * 64 + (T) * 16 + l15];     \
      uint4 blo = brp[hi16];                                             \
      uint4 bhi = brp[hi16 + 2];                                         \
      FR.u[0] = blo.x; FR.u[1] = blo.y; FR.u[2] = blo.z; FR.u[3] = blo.w;\
      FR.u[4] = bhi.x; FR.u[5] = bhi.y; FR.u[6] = bhi.z; FR.u[7] = bhi.w;\
    }
    Frag bf0, bf1, bf2, bf3;
    LOAD_BFRAG(bf0, 0) LOAD_BFRAG(bf1, 1) LOAD_BFRAG(bf2, 2) LOAD_BFRAG(bf3, 3)
#undef LOAD_BFRAG

    acc0 = __builtin_amdgcn_wmma_f32_16x16x32_bf16(false, af.b, false, bf0.b, (short)0, acc0, false, false);
    acc1 = __builtin_amdgcn_wmma_f32_16x16x32_bf16(false, af.b, false, bf1.b, (short)0, acc1, false, false);
    acc2 = __builtin_amdgcn_wmma_f32_16x16x32_bf16(false, af.b, false, bf2.b, (short)0, acc2, false, false);
    acc3 = __builtin_amdgcn_wmma_f32_16x16x32_bf16(false, af.b, false, bf3.b, (short)0, acc3, false, false);
    __syncthreads();
  }

  // C/D layout: lane l, VGPR r -> M = (l>>4)*8 + r, N = l&15
  const int mbase = m0 + wr * 16 + hi16 * 8;
  v8f accs[4] = {acc0, acc1, acc2, acc3};
#pragma unroll
  for (int t = 0; t < 4; ++t) {
    int nc = n0 + wc * 64 + t * 16 + l15;
    if (nc < Nc) {
      float bb = bias ? bias[nc] : 0.f;
#pragma unroll
      for (int r = 0; r < 8; ++r) {
        int gm = mbase + r;
        if (gm < M) {
          float v = accs[t][r] + bb;
          if (relu) v = fmaxf(v, 0.f);
          C[(size_t)gm * Nc + nc] = v;
        }
      }
    }
  }
}

// ---------------------------------------------------------------------------
// Small helper kernels
// ---------------------------------------------------------------------------
__global__ void fill_kernel(float* __restrict__ p, float v, size_t n) {
  size_t i = (size_t)blockIdx.x * blockDim.x + threadIdx.x;
  size_t stride = (size_t)gridDim.x * blockDim.x;
  for (; i < n; i += stride) p[i] = v;
}

// outv[c] = sum_j vec[j] * Wm[j, c]   (1x256 @ 256x256)
__global__ void vecmat256_kernel(const float* __restrict__ vec,
                                 const float* __restrict__ Wm,
                                 float* __restrict__ outv) {
  int c = threadIdx.x;
  float s = 0.f;
  for (int j = 0; j < HH; ++j) s += vec[j] * Wm[j * HH + c];
  outv[c] = s;
}

// ---------------------------------------------------------------------------
// Edge kernels: e-vector rebuilt on the fly from edge_attr (4 scalars) and the
// folded projection M2 = W_ep@We_l [4,256], BV = b_ep@We_l [256].
// Q/K/V head rows are 128B-aligned -> float4 gathers.
// ---------------------------------------------------------------------------
__global__ void edge_alpha_kernel(const int* __restrict__ ei,
                                  const float* __restrict__ eattr,
                                  const float* __restrict__ Q,
                                  const float* __restrict__ Kf,
                                  const float* __restrict__ M2,
                                  const float* __restrict__ BV,
                                  float* __restrict__ alpha,
                                  float* __restrict__ mmax) {
  int t = blockIdx.x * blockDim.x + threadIdx.x;
  if (t >= EE * NHH) return;
  int e  = t >> 3;
  int hh = t & 7;
  int src = ei[e];
  int dst = ei[EE + e];
  float a0 = eattr[e * 4 + 0], a1 = eattr[e * 4 + 1];
  float a2 = eattr[e * 4 + 2], a3 = eattr[e * 4 + 3];
  const int c0 = hh * DD;
  const float4* q4 = (const float4*)(Q  + (size_t)dst * HH + c0);
  const float4* k4 = (const float4*)(Kf + (size_t)src * HH + c0);
  const float4* m04 = (const float4*)(M2 + c0);
  const float4* m14 = (const float4*)(M2 + HH + c0);
  const float4* m24 = (const float4*)(M2 + 2 * HH + c0);
  const float4* m34 = (const float4*)(M2 + 3 * HH + c0);
  const float4* bv4 = (const float4*)(BV + c0);
  float s = 0.f;
#pragma unroll
  for (int d = 0; d < DD / 4; ++d) {
    float4 q = q4[d], k = k4[d];
    float4 w0 = m04[d], w1 = m14[d], w2 = m24[d], w3 = m34[d], bb = bv4[d];
    s += q.x * (k.x + bb.x + a0 * w0.x + a1 * w1.x + a2 * w2.x + a3 * w3.x);
    s += q.y * (k.y + bb.y + a0 * w0.y + a1 * w1.y + a2 * w2.y + a3 * w3.y);
    s += q.z * (k.z + bb.z + a0 * w0.z + a1 * w1.z + a2 * w2.z + a3 * w3.z);
    s += q.w * (k.w + bb.w + a0 * w0.w + a1 * w1.w + a2 * w2.w + a3 * w3.w);
  }
  s *= 0.1767766952966369f;  // 1/sqrt(32)
  alpha[t] = s;
  float* addr = &mmax[(size_t)dst * NHH + hh];
  if (s >= 0.f) atomicMax((int*)addr, __float_as_int(s));
  else          atomicMin((unsigned int*)addr, __float_as_uint(s));
}

__global__ void edge_exp_kernel(const int* __restrict__ ei,
                                float* __restrict__ alpha,
                                const float* __restrict__ mmax,
                                float* __restrict__ den) {
  int t = blockIdx.x * blockDim.x + threadIdx.x;
  if (t >= EE * NHH) return;
  int e  = t >> 3;
  int hh = t & 7;
  int dst = ei[EE + e];
  float ex = __expf(alpha[t] - mmax[(size_t)dst * NHH + hh]);
  alpha[t] = ex;
  atomicAdd(&den[(size_t)dst * NHH + hh], ex);
}

__global__ void edge_agg_kernel(const int* __restrict__ ei,
                                const float* __restrict__ eattr,
                                const float* __restrict__ V,
                                const float* __restrict__ M2,
                                const float* __restrict__ BV,
                                const float* __restrict__ alpha,
                                const float* __restrict__ den,
                                float* __restrict__ agg) {
  int t = blockIdx.x * blockDim.x + threadIdx.x;
  if (t >= EE * NHH) return;
  int e  = t >> 3;
  int hh = t & 7;
  int src = ei[e];
  int dst = ei[EE + e];
  float a = alpha[t] / (den[(size_t)dst * NHH + hh] + 1e-16f);
  float a0 = eattr[e * 4 + 0], a1 = eattr[e * 4 + 1];
  float a2 = eattr[e * 4 + 2], a3 = eattr[e * 4 + 3];
  const int c0 = hh * DD;
  const float4* v4  = (const float4*)(V + (size_t)src * HH + c0);
  const float4* m04 = (const float4*)(M2 + c0);
  const float4* m14 = (const float4*)(M2 + HH + c0);
  const float4* m24 = (const float4*)(M2 + 2 * HH + c0);
  const float4* m34 = (const float4*)(M2 + 3 * HH + c0);
  const float4* bv4 = (const float4*)(BV + c0);
  float* op = agg + (size_t)dst * HH + c0;
#pragma unroll
  for (int d = 0; d < DD / 4; ++d) {
    float4 v = v4[d];
    float4 w0 = m04[d], w1 = m14[d], w2 = m24[d], w3 = m34[d], bb = bv4[d];
    atomicAdd(&op[d * 4 + 0], a * (v.x + bb.x + a0 * w0.x + a1 * w1.x + a2 * w2.x + a3 * w3.x));
    atomicAdd(&op[d * 4 + 1], a * (v.y + bb.y + a0 * w0.y + a1 * w1.y + a2 * w2.y + a3 * w3.y));
    atomicAdd(&op[d * 4 + 2], a * (v.z + bb.z + a0 * w0.z + a1 * w1.z + a2 * w2.z + a3 * w3.z));
    atomicAdd(&op[d * 4 + 3], a * (v.w + bb.w + a0 * w0.w + a1 * w1.w + a2 * w2.w + a3 * w3.w));
  }
}

// ---------------------------------------------------------------------------
// beta gate + mix + LayerNorm + relu + residual  (one block per node)
// ---------------------------------------------------------------------------
__global__ __launch_bounds__(HH)
void combine_kernel(const float* __restrict__ agg, const float* __restrict__ xrb,
                    const float* __restrict__ Wb, const float* __restrict__ g,
                    const float* __restrict__ b, const float* __restrict__ h_in,
                    float* __restrict__ h_out) {
  __shared__ float red[HH];
  int n = blockIdx.x, c = threadIdx.x;
  size_t base = (size_t)n * HH;
  float o  = agg[base + c];
  float xr = xrb[base + c];

  red[c] = Wb[c] * o + Wb[HH + c] * xr + Wb[2 * HH + c] * (o - xr);
  __syncthreads();
  for (int s = HH / 2; s > 0; s >>= 1) { if (c < s) red[c] += red[c + s]; __syncthreads(); }
  float beta = 1.f / (1.f + __expf(-red[0]));
  __syncthreads();

  float hv = beta * xr + (1.f - beta) * o;
  red[c] = hv; __syncthreads();
  for (int s = HH / 2; s > 0; s >>= 1) { if (c < s) red[c] += red[c + s]; __syncthreads(); }
  float mu = red[0] * (1.f / HH);
  __syncthreads();
  float dlt = hv - mu;
  red[c] = dlt * dlt; __syncthreads();
  for (int s = HH / 2; s > 0; s >>= 1) { if (c < s) red[c] += red[c + s]; __syncthreads(); }
  float var = red[0] * (1.f / HH);
  float y = dlt * rsqrtf(var + 1e-5f) * g[c] + b[c];
  h_out[base + c] = fmaxf(y, 0.f) + h_in[base + c];
}

// LN + relu, in place, one block per row of 256
__global__ __launch_bounds__(HH)
void ln_relu_kernel(float* __restrict__ X, const float* __restrict__ g,
                    const float* __restrict__ b) {
  __shared__ float red[HH];
  int n = blockIdx.x, c = threadIdx.x;
  float x = X[(size_t)n * HH + c];
  red[c] = x; __syncthreads();
  for (int s = HH / 2; s > 0; s >>= 1) { if (c < s) red[c] += red[c + s]; __syncthreads(); }
  float mu = red[0] * (1.f / HH);
  __syncthreads();
  float dlt = x - mu;
  red[c] = dlt * dlt; __syncthreads();
  for (int s = HH / 2; s > 0; s >>= 1) { if (c < s) red[c] += red[c + s]; __syncthreads(); }
  float var = red[0] * (1.f / HH);
  X[(size_t)n * HH + c] = fmaxf(dlt * rsqrtf(var + 1e-5f) * g[c] + b[c], 0.f);
}

// ---------------------------------------------------------------------------
// Pooling + head tail
// ---------------------------------------------------------------------------
__global__ void pool_scatter_kernel(const float* __restrict__ hfin,
                                    const int* __restrict__ batch,
                                    float* __restrict__ pool,
                                    float* __restrict__ cnt) {
  int n = blockIdx.x, c = threadIdx.x;
  int gidx = batch[n];
  atomicAdd(&pool[(size_t)gidx * (2 * HH) + HH + c], hfin[(size_t)n * HH + c]);
  if (c == 0) atomicAdd(&cnt[gidx], 1.f);
}

__global__ void pool_finalize_kernel(float* __restrict__ pool,
                                     const float* __restrict__ cnt) {
  int gidx = blockIdx.x, c = threadIdx.x;
  float add = pool[(size_t)gidx * (2 * HH) + HH + c];
  pool[(size_t)gidx * (2 * HH) + c] = add / fmaxf(cnt[gidx], 1.f);
}

__global__ void head_out_kernel(const float* __restrict__ C2,
                                const float* __restrict__ Wc3,
                                const float* __restrict__ bc3,
                                float* __restrict__ outp) {
  int gidx = blockIdx.x * blockDim.x + threadIdx.x;
  if (gidx >= GG) return;
  float s = bc3[0];
  for (int j = 0; j < HH / 2; ++j) s += C2[gidx * (HH / 2) + j] * Wc3[j];
  outp[gidx] = s;
}

// ---------------------------------------------------------------------------
static void launch_gemm(const float* A, const float* B, const float* bias,
                        float* C, int M, int Nc, int Kc, int relu,
                        hipStream_t s) {
  dim3 grid((Nc + TILE_N - 1) / TILE_N, (M + TILE_M - 1) / TILE_M);
  gemm_wmma_bf16<<<grid, 256, 0, s>>>(A, B, bias, C, M, Nc, Kc, relu);
}

extern "C" void kernel_launch(void* const* d_in, const int* in_sizes, int n_in,
                              void* d_out, int out_size, void* d_ws, size_t ws_size,
                              hipStream_t stream) {
  const float* x      = (const float*)d_in[0];
  const int*   ei     = (const int*)  d_in[1];
  const float* eattr  = (const float*)d_in[2];
  const int*   batch  = (const int*)  d_in[3];
  const float* W_in   = (const float*)d_in[4];
  const float* b_in   = (const float*)d_in[5];
  const float* W_ep   = (const float*)d_in[6];
  const float* b_ep   = (const float*)d_in[7];
  const float* Wq     = (const float*)d_in[8];
  const float* bq     = (const float*)d_in[9];
  const float* Wk     = (const float*)d_in[10];
  const float* bk     = (const float*)d_in[11];
  const float* Wv     = (const float*)d_in[12];
  const float* bv     = (const float*)d_in[13];
  const float* We     = (const float*)d_in[14];
  const float* Wskip  = (const float*)d_in[15];
  const float* bskip  = (const float*)d_in[16];
  const float* Wbeta  = (const float*)d_in[17];
  const float* ln_g   = (const float*)d_in[18];
  const float* ln_b   = (const float*)d_in[19];
  const float* Wc1    = (const float*)d_in[20];
  const float* bc1    = (const float*)d_in[21];
  const float* lnc_g  = (const float*)d_in[22];
  const float* lnc_b  = (const float*)d_in[23];
  const float* Wc2    = (const float*)d_in[24];
  const float* bc2    = (const float*)d_in[25];
  const float* Wc3    = (const float*)d_in[26];
  const float* bc3    = (const float*)d_in[27];

  // ---- workspace layout (floats) ----
  float* W = (float*)d_ws;
  const size_t NHf = (size_t)NN * HH;
  float* hA    = W;
  float* hB    = W + NHf;
  float* Qb    = W + 2 * NHf;
  float* Kb    = W + 3 * NHf;
  float* Vb    = W + 4 * NHf;
  float* XRb   = W + 5 * NHf;
  float* AGGb  = W + 6 * NHf;
  float* ALPHA = W + 7 * NHf;                     // EE*NHH
  float* MMAX  = ALPHA + (size_t)EE * NHH;        // NN*NHH
  float* DEN   = MMAX  + (size_t)NN * NHH;        // NN*NHH
  float* M2b   = DEN   + (size_t)NN * NHH;        // EDIM*HH
  float* BVb   = M2b   + EDIM * HH;               // HH
  float* POOL  = BVb   + HH;                      // GG*2*HH
  float* CNT   = POOL  + (size_t)GG * 2 * HH;     // GG
  float* C1b   = CNT   + GG;                      // GG*HH
  float* C2b   = C1b   + (size_t)GG * HH;         // GG*HH/2

  const int eth = EE * NHH;
  const int eblocks = (eth + 255) / 256;

  // h = x @ W_in + b_in
  launch_gemm(x, W_in, b_in, hA, NN, HH, IN_F, 0, stream);

  for (int l = 0; l < LL; ++l) {
    const float* We_l = We + (size_t)l * HH * HH;
    // fold edge projection: M2 = W_ep @ We_l, BV = b_ep @ We_l
    launch_gemm(W_ep, We_l, nullptr, M2b, EDIM, HH, HH, 0, stream);
    vecmat256_kernel<<<1, HH, 0, stream>>>(b_ep, We_l, BVb);

    launch_gemm(hA, Wq    + (size_t)l * HH * HH, bq    + l * HH, Qb,  NN, HH, HH, 0, stream);
    launch_gemm(hA, Wk    + (size_t)l * HH * HH, bk    + l * HH, Kb,  NN, HH, HH, 0, stream);
    launch_gemm(hA, Wv    + (size_t)l * HH * HH, bv    + l * HH, Vb,  NN, HH, HH, 0, stream);
    launch_gemm(hA, Wskip + (size_t)l * HH * HH, bskip + l * HH, XRb, NN, HH, HH, 0, stream);

    fill_kernel<<<1024, 256, 0, stream>>>(AGGb, 0.f, NHf);
    fill_kernel<<<256, 256, 0, stream>>>(MMAX, -INFINITY, (size_t)NN * NHH);
    fill_kernel<<<256, 256, 0, stream>>>(DEN, 0.f, (size_t)NN * NHH);

    edge_alpha_kernel<<<eblocks, 256, 0, stream>>>(ei, eattr, Qb, Kb, M2b, BVb, ALPHA, MMAX);
    edge_exp_kernel  <<<eblocks, 256, 0, stream>>>(ei, ALPHA, MMAX, DEN);
    edge_agg_kernel  <<<eblocks, 256, 0, stream>>>(ei, eattr, Vb, M2b, BVb, ALPHA, DEN, AGGb);

    combine_kernel<<<NN, HH, 0, stream>>>(AGGb, XRb, Wbeta + (size_t)l * 3 * HH,
                                          ln_g + l * HH, ln_b + l * HH, hA, hB);
    float* tmp = hA; hA = hB; hB = tmp;
  }

  // pooling: ge = [add/cnt, add]
  fill_kernel<<<64, 256, 0, stream>>>(POOL, 0.f, (size_t)GG * 2 * HH);
  fill_kernel<<<1, 64, 0, stream>>>(CNT, 0.f, (size_t)GG);
  pool_scatter_kernel <<<NN, HH, 0, stream>>>(hA, batch, POOL, CNT);
  pool_finalize_kernel<<<GG, HH, 0, stream>>>(POOL, CNT);

  // head MLP
  launch_gemm(POOL, Wc1, bc1, C1b, GG, HH, 2 * HH, 0, stream);
  ln_relu_kernel<<<GG, HH, 0, stream>>>(C1b, lnc_g, lnc_b);
  launch_gemm(C1b, Wc2, bc2, C2b, GG, HH / 2, HH, 1, stream);
  head_out_kernel<<<1, 64, 0, stream>>>(C2b, Wc3, bc3, (float*)d_out);
}
